// LSTMAttnDecoder_66005057405751
// MI455X (gfx1250) — matrix operand order repaired and checked by reading, not compiled, not run on hardware
//
#include <hip/hip_runtime.h>
#include <hip/hip_bf16.h>
#include <math.h>

// ---------------- constants (match reference) ----------------
#define Hh   1024
#define Vv   32000
#define Bb   16
#define Ss   256
#define Tt   200
#define SOS  1
#define G4   4096   // 4*H

typedef __bf16 bf_t;
typedef __attribute__((ext_vector_type(16))) __bf16  v16bf;
typedef __attribute__((ext_vector_type(8)))  __bf16  v8bf;
typedef __attribute__((ext_vector_type(8)))  float   v8f;

// ---------------- fp32 -> bf16 conversion ----------------
__global__ void f32_to_bf16_kernel(const float* __restrict__ in, bf_t* __restrict__ out, size_t n) {
    size_t i = (size_t)blockIdx.x * blockDim.x + threadIdx.x;
    size_t stride = (size_t)gridDim.x * blockDim.x;
    for (; i < n; i += stride) out[i] = (bf_t)in[i];
}

// ---------------- generic 16xN bf16 WMMA GEMM, 2 N-tiles per wave ----------------
// Cout[mbase..mbase+15, :] = A0*W0^T (+ A1*W1^T) + bias0 + bias1
// A row-major (M x K), W row-major (N x K). Each wave owns two adjacent 16-wide
// N tiles (N/16 must be even), so every A fragment feeds two unconditional
// v_wmma_f32_16x16x32_bf16 — straight-line body keeps the loads pipelined.
__global__ void gemm16_bf16_kernel(
    const bf_t* __restrict__ A0, const bf_t* __restrict__ W0, int K0,
    const bf_t* __restrict__ A1, const bf_t* __restrict__ W1, int K1,
    const float* __restrict__ bias0, const float* __restrict__ bias1,
    float* __restrict__ Cout, int N, int ldc)
{
    int wave   = threadIdx.x >> 5;
    int lane   = threadIdx.x & 31;
    int pair   = blockIdx.x * (blockDim.x >> 5) + wave;
    int ntiles = N >> 4;
    int ntile0 = pair * 2;
    if (ntile0 >= ntiles) return;
    int mbase  = blockIdx.y * 16;

    int mrow  = lane & 15;
    int aksel = (lane < 16) ? 0 : 16;   // A-fragment K sub-offset (contiguous, matches B)
    int bksel = (lane < 16) ? 0 : 16;   // B-fragment K sub-offset (contiguous, matches A)
    int ncol0 = (ntile0 << 4) + (lane & 15);

    v8f acc0 = {}, acc1 = {};

    { // segment 0
        const bf_t* arow  = A0 + (size_t)(mbase + mrow) * K0;
        const bf_t* b0row = W0 + (size_t)ncol0 * K0;
        const bf_t* b1row = b0row + (size_t)16 * K0;     // adjacent N tile
        for (int k = 0; k < K0; k += 32) {
            v8bf alo = *(const v8bf*)(arow + k + aksel);        // K+0..7   (or 16..23)
            v8bf ahi = *(const v8bf*)(arow + k + aksel + 8);    // K+8..15  (or 24..31)
            v16bf a;
            #pragma unroll
            for (int i = 0; i < 8; ++i) { a[i] = alo[i]; a[8 + i] = ahi[i]; }
            v16bf b0 = *(const v16bf*)(b0row + k + bksel);
            v16bf b1 = *(const v16bf*)(b1row + k + bksel);
            acc0 = __builtin_amdgcn_wmma_f32_16x16x32_bf16(
                false, a, false, b0, (short)0, acc0, false, false);
            acc1 = __builtin_amdgcn_wmma_f32_16x16x32_bf16(
                false, a, false, b1, (short)0, acc1, false, false);
        }
    }
    if (A1) { // optional segment 1 (e.g. recurrent term of LSTM gates)
        const bf_t* arow  = A1 + (size_t)(mbase + mrow) * K1;
        const bf_t* b0row = W1 + (size_t)ncol0 * K1;
        const bf_t* b1row = b0row + (size_t)16 * K1;
        for (int k = 0; k < K1; k += 32) {
            v8bf alo = *(const v8bf*)(arow + k + aksel);
            v8bf ahi = *(const v8bf*)(arow + k + aksel + 8);
            v16bf a;
            #pragma unroll
            for (int i = 0; i < 8; ++i) { a[i] = alo[i]; a[8 + i] = ahi[i]; }
            v16bf b0 = *(const v16bf*)(b0row + k + bksel);
            v16bf b1 = *(const v16bf*)(b1row + k + bksel);
            acc0 = __builtin_amdgcn_wmma_f32_16x16x32_bf16(
                false, a, false, b0, (short)0, acc0, false, false);
            acc1 = __builtin_amdgcn_wmma_f32_16x16x32_bf16(
                false, a, false, b1, (short)0, acc1, false, false);
        }
    }

    // D layout: VGPR r -> M=r (lanes 0-15) / M=8+r (lanes 16-31), N = lane&15
    int ncol1 = ncol0 + 16;
    float biasA = 0.f, biasB = 0.f;
    if (bias0) { biasA += bias0[ncol0]; biasB += bias0[ncol1]; }
    if (bias1) { biasA += bias1[ncol0]; biasB += bias1[ncol1]; }
    #pragma unroll
    for (int r = 0; r < 8; ++r) {
        int m = mbase + ((lane < 16) ? r : r + 8);
        Cout[(size_t)m * ldc + ncol0] = acc0[r] + biasA;
        Cout[(size_t)m * ldc + ncol1] = acc1[r] + biasB;
    }
}

// ---------------- attention scores: tanh(qWa + keys_proj) . Va ----------------
__global__ void attn_scores_kernel(const float* __restrict__ keys_proj,
                                   const float* __restrict__ qWa,
                                   const float* __restrict__ Va_w,
                                   const float* __restrict__ Va_b,
                                   float* __restrict__ scores)
{
    int gwave = (blockIdx.x * blockDim.x + threadIdx.x) >> 5;   // one wave per (b,s)
    int lane  = threadIdx.x & 31;
    if (gwave >= Bb * Ss) return;
    int b = gwave >> 8;                                          // S == 256
    const float* kp = keys_proj + (size_t)gwave * Hh;
    const float* qw = qWa + (size_t)b * Hh;
    float p = 0.f;
    for (int h = lane; h < Hh; h += 32)
        p += Va_w[h] * tanhf(qw[h] + kp[h]);
    #pragma unroll
    for (int off = 16; off; off >>= 1) p += __shfl_xor(p, off, 32);
    if (lane == 0) scores[gwave] = p + Va_b[0];
}

// ---------------- softmax over S + context vector ----------------
__global__ void softmax_ctx_kernel(const float* __restrict__ scores,
                                   const float* __restrict__ enc,       // (B,S,H) fp32
                                   float* __restrict__ attn_out,        // (B,T,S) region
                                   float* __restrict__ ctx, int t)
{
    __shared__ float sAttn[Ss];
    __shared__ float red[Ss];
    int b = blockIdx.x, tid = threadIdx.x;
    float v = scores[b * Ss + tid];
    red[tid] = v; __syncthreads();
    for (int o = 128; o; o >>= 1) { if (tid < o) red[tid] = fmaxf(red[tid], red[tid + o]); __syncthreads(); }
    float m = red[0]; __syncthreads();
    float e = expf(v - m);
    red[tid] = e; __syncthreads();
    for (int o = 128; o; o >>= 1) { if (tid < o) red[tid] += red[tid + o]; __syncthreads(); }
    float a = e / red[0];
    sAttn[tid] = a;
    attn_out[((size_t)b * Tt + t) * Ss + tid] = a;
    __syncthreads();
    #pragma unroll
    for (int j = 0; j < 4; ++j) {
        int h = tid + j * 256;
        float acc = 0.f;
        for (int s = 0; s < Ss; ++s)
            acc += sAttn[s] * enc[((size_t)b * Ss + s) * Hh + h];
        ctx[b * Hh + h] = acc;
    }
}

// ---------------- x = [embedding[tok] , ctx] in bf16 ----------------
__global__ void build_x_kernel(const int* __restrict__ target,
                               const float* __restrict__ embedding,
                               const float* __restrict__ ctx,
                               bf_t* __restrict__ x, int t)
{
    int b = blockIdx.x, tid = threadIdx.x;
    int tok = (t == 0) ? SOS : target[b * Tt + (t - 1)];
    const float* erow = embedding + (size_t)tok * Hh;
    for (int h = tid; h < Hh; h += blockDim.x) {
        x[(size_t)b * 2 * Hh + h]       = (bf_t)erow[h];
        x[(size_t)b * 2 * Hh + Hh + h]  = (bf_t)ctx[b * Hh + h];
    }
}

// ---------------- LSTM activation: h = sig(o)*tanh(sig(i)*tanh(g)) ----------------
__global__ void lstm_act_kernel(const float* __restrict__ gates,
                                float* __restrict__ h, bf_t* __restrict__ hb)
{
    int idx = blockIdx.x * blockDim.x + threadIdx.x;   // B*H
    int b = idx >> 10, hh = idx & 1023;
    const float* g = gates + (size_t)b * G4;
    float gi = g[hh], gg = g[2 * Hh + hh], go = g[3 * Hh + hh];
    float c  = (1.f / (1.f + expf(-gi))) * tanhf(gg);  // f*c_prev = 0 per reference
    float hv = (1.f / (1.f + expf(-go))) * tanhf(c);
    h[idx]  = hv;
    hb[idx] = (bf_t)hv;
}

// ---------------- log_softmax over V, write output row ----------------
__global__ void logsoftmax_kernel(const float* __restrict__ logits,
                                  float* __restrict__ out, int t)
{
    __shared__ float red[1024];
    int b = blockIdx.x, tid = threadIdx.x;
    const float* row = logits + (size_t)b * Vv;
    float m = -INFINITY;
    for (int i = tid; i < Vv; i += 1024) m = fmaxf(m, row[i]);
    red[tid] = m; __syncthreads();
    for (int o = 512; o; o >>= 1) { if (tid < o) red[tid] = fmaxf(red[tid], red[tid + o]); __syncthreads(); }
    m = red[0]; __syncthreads();
    float s = 0.f;
    for (int i = tid; i < Vv; i += 1024) s += expf(row[i] - m);
    red[tid] = s; __syncthreads();
    for (int o = 512; o; o >>= 1) { if (tid < o) red[tid] += red[tid + o]; __syncthreads(); }
    float lse = m + logf(red[0]);
    float* orow = out + ((size_t)b * Tt + t) * Vv;
    for (int i = tid; i < Vv; i += 1024) orow[i] = row[i] - lse;
}

// ---------------- init / finalize hidden state ----------------
__global__ void init_h_kernel(const float* __restrict__ enc_hidden,
                              float* __restrict__ h0, float* __restrict__ h1,
                              bf_t* __restrict__ h0b, bf_t* __restrict__ h1b)
{
    int idx = blockIdx.x * blockDim.x + threadIdx.x;   // B*H
    float a = enc_hidden[idx];
    float c = enc_hidden[Bb * Hh + idx];
    h0[idx] = a; h1[idx] = c;
    h0b[idx] = (bf_t)a; h1b[idx] = (bf_t)c;
}

__global__ void write_hfinal_kernel(const float* __restrict__ h0,
                                    const float* __restrict__ h1,
                                    float* __restrict__ out)
{
    int idx = blockIdx.x * blockDim.x + threadIdx.x;   // B*H
    out[idx] = h0[idx];
    out[Bb * Hh + idx] = h1[idx];
}

// ---------------- host side ----------------
static inline size_t alignup256(size_t x) { return (x + 255) & ~(size_t)255; }

extern "C" void kernel_launch(void* const* d_in, const int* in_sizes, int n_in,
                              void* d_out, int out_size, void* d_ws, size_t ws_size,
                              hipStream_t stream) {
    (void)in_sizes; (void)n_in; (void)out_size; (void)ws_size;

    const float* enc      = (const float*)d_in[0];   // (B,S,H)
    const float* enc_hid  = (const float*)d_in[1];   // (L,B,H)
    const int*   target   = (const int*)  d_in[2];   // (B,T)
    const float* emb      = (const float*)d_in[3];   // (V,H)
    const float* Wa_w     = (const float*)d_in[4];
    const float* Wa_b     = (const float*)d_in[5];
    const float* Ua_w     = (const float*)d_in[6];
    const float* Ua_b     = (const float*)d_in[7];
    const float* Va_w     = (const float*)d_in[8];
    const float* Va_b     = (const float*)d_in[9];
    const float* W_ih0    = (const float*)d_in[10];  // (4H,2H)
    const float* W_hh0    = (const float*)d_in[11];  // (4H,H)
    const float* b_ih0    = (const float*)d_in[12];
    const float* b_hh0    = (const float*)d_in[13];
    const float* W_ih1    = (const float*)d_in[14];  // (4H,H)
    const float* W_hh1    = (const float*)d_in[15];
    const float* b_ih1    = (const float*)d_in[16];
    const float* b_hh1    = (const float*)d_in[17];
    const float* out_w    = (const float*)d_in[18];  // (V,H)
    const float* out_b    = (const float*)d_in[19];

    float* logp  = (float*)d_out;                                    // (B,T,V)
    float* hfin  = logp + (size_t)Bb * Tt * Vv;                      // (L,B,H)
    float* attns = hfin + (size_t)2 * Bb * Hh;                       // (B,T,S)

    // ---- workspace carve-out (~140 MB) ----
    char* w = (char*)d_ws;
    auto alloc = [&](size_t bytes) { char* p = w; w += alignup256(bytes); return p; };
    bf_t* outw_b  = (bf_t*)alloc((size_t)Vv * Hh * 2);
    bf_t* Wih0_b  = (bf_t*)alloc((size_t)G4 * 2 * Hh * 2);
    bf_t* Whh0_b  = (bf_t*)alloc((size_t)G4 * Hh * 2);
    bf_t* Wih1_b  = (bf_t*)alloc((size_t)G4 * Hh * 2);
    bf_t* Whh1_b  = (bf_t*)alloc((size_t)G4 * Hh * 2);
    bf_t* Wa_b16  = (bf_t*)alloc((size_t)Hh * Hh * 2);
    bf_t* Ua_b16  = (bf_t*)alloc((size_t)Hh * Hh * 2);
    bf_t* enc_b   = (bf_t*)alloc((size_t)Bb * Ss * Hh * 2);
    float* keysp  = (float*)alloc((size_t)Bb * Ss * Hh * 4);
    float* qWa    = (float*)alloc((size_t)Bb * Hh * 4);
    float* scores = (float*)alloc((size_t)Bb * Ss * 4);
    float* ctx    = (float*)alloc((size_t)Bb * Hh * 4);
    bf_t*  xbuf   = (bf_t*) alloc((size_t)Bb * 2 * Hh * 2);
    float* gates  = (float*)alloc((size_t)Bb * G4 * 4);
    float* h0     = (float*)alloc((size_t)Bb * Hh * 4);
    float* h1     = (float*)alloc((size_t)Bb * Hh * 4);
    bf_t*  h0b    = (bf_t*) alloc((size_t)Bb * Hh * 2);
    bf_t*  h1b    = (bf_t*) alloc((size_t)Bb * Hh * 2);
    float* logits = (float*)alloc((size_t)Bb * Vv * 4);

    // ---- one-time weight conversion fp32 -> bf16 ----
    auto conv = [&](const float* src, bf_t* dst, size_t n) {
        int blocks = (int)((n + 255 * 8) / (256 * 8));
        if (blocks > 4096) blocks = 4096;
        f32_to_bf16_kernel<<<blocks, 256, 0, stream>>>(src, dst, n);
    };
    conv(out_w, outw_b, (size_t)Vv * Hh);
    conv(W_ih0, Wih0_b, (size_t)G4 * 2 * Hh);
    conv(W_hh0, Whh0_b, (size_t)G4 * Hh);
    conv(W_ih1, Wih1_b, (size_t)G4 * Hh);
    conv(W_hh1, Whh1_b, (size_t)G4 * Hh);
    conv(Wa_w,  Wa_b16, (size_t)Hh * Hh);
    conv(Ua_w,  Ua_b16, (size_t)Hh * Hh);
    conv(enc,   enc_b,  (size_t)Bb * Ss * Hh);

    init_h_kernel<<<64, 256, 0, stream>>>(enc_hid, h0, h1, h0b, h1b);

    // grid helper: pairs of N tiles per wave, 8 waves per block (N/16 even)
    auto nblocks = [](int N) { return ((N >> 4) / 2 + 7) / 8; };

    // keys_proj = enc @ Ua^T + Ua_b : M = B*S = 4096 (grid.y = 256), N = K = 1024
    gemm16_bf16_kernel<<<dim3(nblocks(Hh), 256), 256, 0, stream>>>(
        enc_b, Ua_b16, Hh, nullptr, nullptr, 0, Ua_b, nullptr, keysp, Hh, Hh);

    // ---- 200 sequential decoder steps ----
    for (int t = 0; t < Tt; ++t) {
        // qWa = h1_prev @ Wa^T + Wa_b      (16 x 1024, K=1024)
        gemm16_bf16_kernel<<<dim3(nblocks(Hh), 1), 256, 0, stream>>>(
            h1b, Wa_b16, Hh, nullptr, nullptr, 0, Wa_b, nullptr, qWa, Hh, Hh);
        // scores (one wave per (b,s))
        attn_scores_kernel<<<(Bb * Ss * 32) / 256, 256, 0, stream>>>(
            keysp, qWa, Va_w, Va_b, scores);
        // softmax + context + write attention output
        softmax_ctx_kernel<<<Bb, 256, 0, stream>>>(scores, enc, attns, ctx, t);
        // x = [emb(tok) , ctx] bf16
        build_x_kernel<<<Bb, 256, 0, stream>>>(target, emb, ctx, xbuf, t);
        // LSTM layer 0 gates: x@W_ih0^T + h0@W_hh0^T + biases  (N=4096)
        gemm16_bf16_kernel<<<dim3(nblocks(G4), 1), 256, 0, stream>>>(
            xbuf, Wih0_b, 2 * Hh, h0b, Whh0_b, Hh, b_ih0, b_hh0, gates, G4, G4);
        lstm_act_kernel<<<64, 256, 0, stream>>>(gates, h0, h0b);
        // LSTM layer 1 gates: h0_new@W_ih1^T + h1_prev@W_hh1^T + biases
        gemm16_bf16_kernel<<<dim3(nblocks(G4), 1), 256, 0, stream>>>(
            h0b, Wih1_b, Hh, h1b, Whh1_b, Hh, b_ih1, b_hh1, gates, G4, G4);
        lstm_act_kernel<<<64, 256, 0, stream>>>(gates, h1, h1b);
        // logits = h1_new @ out_w^T + out_b   (N=32000, K=1024) -> 1000 wave-pairs
        gemm16_bf16_kernel<<<dim3(nblocks(Vv), 1), 256, 0, stream>>>(
            h1b, outw_b, Hh, nullptr, nullptr, 0, out_b, nullptr, logits, Vv, Vv);
        // log_softmax row-wise into output
        logsoftmax_kernel<<<Bb, 1024, 0, stream>>>(logits, logp, t);
    }

    write_hfinal_kernel<<<64, 256, 0, stream>>>(h0, h1, hfin);
}
